// FusionMagic_5411658793230
// MI455X (gfx1250) — matrix operand order repaired and checked by reading, not compiled, not run on hardware
//
#include <hip/hip_runtime.h>
#include <hip/hip_bf16.h>
#include <math.h>

typedef float    f32x4 __attribute__((ext_vector_type(4)));
typedef __attribute__((ext_vector_type(16))) _Float16 v16h;
typedef __attribute__((ext_vector_type(8)))  float    v8f;

#define RED_THREADS 256

// ---------------------------------------------------------------------------
// Pass 1: streaming partial reduction. grid = (rows, chunks); each block sums
// a contiguous chunk of one (batch,channel) row with b128 non-temporal loads
// and writes ONE partial (no atomics -> bitwise deterministic across replays).
// ---------------------------------------------------------------------------
__global__ __launch_bounds__(RED_THREADS) void fm_reduce_partial(
    const float* __restrict__ x, float* __restrict__ part, int spatial, int chunks)
{
  const int row   = blockIdx.x;
  const int chunk = blockIdx.y;
  const int cs    = spatial / chunks;               // elems per chunk (mult. of 1024)
  const f32x4* __restrict__ p =
      (const f32x4*)(x + (size_t)row * (size_t)spatial + (size_t)chunk * (size_t)cs);
  const int n4 = cs >> 2;
  float s = 0.f;
  for (int i = threadIdx.x; i < n4; i += RED_THREADS) {
    f32x4 v = __builtin_nontemporal_load(&p[i]);    // no-reuse stream: NT hint
    s += (v.x + v.y) + (v.z + v.w);
  }
  __shared__ float red[RED_THREADS];
  red[threadIdx.x] = s;
  __syncthreads();
  for (int t = RED_THREADS / 2; t > 0; t >>= 1) {
    if (threadIdx.x < t) red[threadIdx.x] += red[threadIdx.x + t];
    __syncthreads();
  }
  if (threadIdx.x == 0) part[row * chunks + chunk] = red[0];
}

// ---------------------------------------------------------------------------
// Fused tail: combine partials -> means -> LN chain -> 3 WMMA GEMMs -> sigmoid
// ---------------------------------------------------------------------------
__device__ __forceinline__ float fm_block_sum(float v, float* s_red) {
  const int tid = threadIdx.x;
  s_red[tid] = v;
  __syncthreads();
  for (int t = 128; t > 0; t >>= 1) {
    if (tid < t) s_red[tid] += s_red[tid + t];
    __syncthreads();
  }
  float r = s_red[0];
  __syncthreads();
  return r;
}

__device__ void fm_ln_inplace(float* v, int n, const float* __restrict__ w,
                              const float* __restrict__ b, float* s_red) {
  const int tid = threadIdx.x;
  float s = 0.f;
  for (int i = tid; i < n; i += 256) s += v[i];
  const float mu = fm_block_sum(s, s_red) / (float)n;
  float q = 0.f;
  for (int i = tid; i < n; i += 256) { float d = v[i] - mu; q += d * d; }
  const float var = fm_block_sum(q, s_red) / (float)n;
  const float r = rsqrtf(var + 1e-5f);
  for (int i = tid; i < n; i += 256) v[i] = (v[i] - mu) * r * w[i] + b[i];
  __syncthreads();
}

// One 16x16 D tile of A[16 x Ktot] * B[Ktot x 16], B[k][n] = W[n*ldw + k].
// A lives in LDS as f16, rows 0..1 valid (M padded with zeros).
// A/B in contiguous k order (k = 16*(lane/16) + s); D layout per CDNA5 ISA 7.12.2 (wave32).
__device__ __forceinline__ v8f fm_tile_mm(const _Float16* __restrict__ sA, int lda,
                                          int Ktot, const float* __restrict__ W,
                                          int ldw, int n0, int lane) {
  v8f acc = {0.f, 0.f, 0.f, 0.f, 0.f, 0.f, 0.f, 0.f};
  const int  m  = lane & 15;
  const bool hi = lane >= 16;
  const float* __restrict__ wrow = W + (size_t)(n0 + (lane & 15)) * (size_t)ldw;
  for (int kb = 0; kb < Ktot; kb += 32) {
    v16h a, bf;
    const int ka = kb + (hi ? 16 : 0);             // A: lanes16-31 hold K+16..K+31
    if (m < 2) {
      const _Float16* ap = sA + m * lda;
#pragma unroll
      for (int h = 0; h < 8; ++h) { a[h] = ap[ka + h]; a[8 + h] = ap[ka + 8 + h]; }
    } else {
#pragma unroll
      for (int h = 0; h < 16; ++h) a[h] = (_Float16)0.f;
    }
    const int kbB = kb + (hi ? 16 : 0);            // B: lanes16-31 hold K+16..K+31
#pragma unroll
    for (int h = 0; h < 16; ++h) bf[h] = (_Float16)wrow[kbB + h];
    acc = __builtin_amdgcn_wmma_f32_16x16x32_f16(false, a, false, bf, (short)0, acc,
                                                 false, false);
  }
  return acc;
}

__global__ __launch_bounds__(256) void fm_finalize(
    const float* __restrict__ ws,
    const float* __restrict__ ln1w, const float* __restrict__ ln1b,
    const float* __restrict__ ln2w, const float* __restrict__ ln2b,
    const float* __restrict__ ln3w, const float* __restrict__ ln3b,
    const float* __restrict__ ln4w, const float* __restrict__ ln4b,
    const float* __restrict__ ln5w, const float* __restrict__ ln5b,
    const float* __restrict__ w1,   const float* __restrict__ b1,
    const float* __restrict__ w2,   const float* __restrict__ b2,
    const float* __restrict__ cw,   const float* __restrict__ cb,
    float* __restrict__ out)
{
  __shared__ float    s_u[2 * 448];   // working f32 vector (out7 layout)
  __shared__ float    s_red[256];
  __shared__ _Float16 s_a[2 * 448];   // f16 A operand for WMMA
  __shared__ float    s_h[2 * 448];   // GEMM outputs (h1 then h2)
  const int tid = threadIdx.x;

  // 1) partials -> means, placed directly in out7 order:
  //    [b*448 + 0..63]=mean(x2)  [64..191]=mean(x1)  [192..447]=mean(x0)
  for (int i = tid; i < 896; i += 256) {
    float s = 0.f;
    if (i < 128) {                               // x2: 128 rows x 128 chunks @ ws+0
      const float* q = ws + i * 128;
      for (int j = 0; j < 128; ++j) s += q[j];
      s_u[(i >> 6) * 448 + (i & 63)] = s * (1.f / 2097152.f);
    } else if (i < 384) {                        // x1: 256 rows x 32 chunks @ ws+16384
      const int r = i - 128;
      const float* q = ws + 16384 + r * 32;
      for (int j = 0; j < 32; ++j) s += q[j];
      s_u[(r >> 7) * 448 + 64 + (r & 127)] = s * (1.f / 262144.f);
    } else {                                     // x0: 512 rows x 8 chunks @ ws+24576
      const int r = i - 384;
      const float* q = ws + 24576 + r * 8;
      for (int j = 0; j < 8; ++j) s += q[j];
      s_u[(r >> 8) * 448 + 192 + (r & 255)] = s * (1.f / 32768.f);
    }
  }
  __syncthreads();

  // 2) LayerNorm chain (matches reference nesting exactly)
  for (int b = 0; b < 2; ++b) {
    float* u = s_u + b * 448;
    fm_ln_inplace(u,        64, ln1w, ln1b, s_red);   // v1 = LN(gap(x2))
    fm_ln_inplace(u + 64,  128, ln2w, ln2b, s_red);   // v2 = LN(gap(x1))
    fm_ln_inplace(u + 192, 256, ln3w, ln3b, s_red);   // v3 = LN(gap(x0))
    fm_ln_inplace(u + 64,  384, ln4w, ln4b, s_red);   // LN([v2,v3])
    fm_ln_inplace(u,       448, ln5w, ln5b, s_red);   // LN([v1,out6]) = out7
  }

  // 3) out7 -> f16 A operand
  for (int i = tid; i < 896; i += 256) s_a[i] = (_Float16)s_u[i];
  __syncthreads();

  const int wave = tid >> 5, lane = tid & 31;

  // 4) h1 = relu(out7 @ w1^T + b1) : [2,64], K=448 (4 N-tiles, waves 0..3)
  if (wave < 4) {
    v8f acc = fm_tile_mm(s_a, 448, 448, w1, 448, wave * 16, lane);
    if (lane < 16) {
      const int j = wave * 16 + lane;
      s_h[j]      = fmaxf(acc[0] + b1[j], 0.f);       // D VGPR0 lanes0-15 = row M=0
      s_h[64 + j] = fmaxf(acc[1] + b1[j], 0.f);       // D VGPR1 lanes0-15 = row M=1
    }
  }
  __syncthreads();
  for (int i = tid; i < 128; i += 256) s_a[i] = (_Float16)s_h[i];
  __syncthreads();

  // 5) h2 = relu(h1 @ w2^T + b2) : [2,448], K=64 (28 N-tiles over 8 waves)
  for (int t = wave; t < 28; t += 8) {
    v8f acc = fm_tile_mm(s_a, 64, 64, w2, 64, t * 16, lane);
    if (lane < 16) {
      const int j = t * 16 + lane;
      s_h[j]       = fmaxf(acc[0] + b2[j], 0.f);
      s_h[448 + j] = fmaxf(acc[1] + b2[j], 0.f);
    }
  }
  __syncthreads();
  for (int i = tid; i < 896; i += 256) s_a[i] = (_Float16)s_h[i];
  __syncthreads();

  // 6) o = sigmoid(h2 @ cw^T + cb) : [2,64], K=448
  if (wave < 4) {
    v8f acc = fm_tile_mm(s_a, 448, 448, cw, 448, wave * 16, lane);
    if (lane < 16) {
      const int j = wave * 16 + lane;
      out[j]      = 1.f / (1.f + expf(-(acc[0] + cb[j])));
      out[64 + j] = 1.f / (1.f + expf(-(acc[1] + cb[j])));
    }
  }
}

// ---------------------------------------------------------------------------
extern "C" void kernel_launch(void* const* d_in, const int* in_sizes, int n_in,
                              void* d_out, int out_size, void* d_ws, size_t ws_size,
                              hipStream_t stream) {
  (void)in_sizes; (void)n_in; (void)out_size; (void)ws_size;
  const float* x0   = (const float*)d_in[0];   // [2,256,32^3]
  const float* x1   = (const float*)d_in[1];   // [2,128,64^3]
  const float* x2   = (const float*)d_in[2];   // [2, 64,128^3]
  const float* ln1w = (const float*)d_in[3];
  const float* ln1b = (const float*)d_in[4];
  const float* ln2w = (const float*)d_in[5];
  const float* ln2b = (const float*)d_in[6];
  const float* ln3w = (const float*)d_in[7];
  const float* ln3b = (const float*)d_in[8];
  const float* ln4w = (const float*)d_in[9];
  const float* ln4b = (const float*)d_in[10];
  const float* ln5w = (const float*)d_in[11];
  const float* ln5b = (const float*)d_in[12];
  const float* w1   = (const float*)d_in[13];  // [64,448]
  const float* b1   = (const float*)d_in[14];
  const float* w2   = (const float*)d_in[15];  // [448,64]
  const float* b2   = (const float*)d_in[16];
  const float* cw   = (const float*)d_in[17];  // [64,448]
  const float* cb   = (const float*)d_in[18];
  float* ws  = (float*)d_ws;                   // 28672 floats of partials
  float* out = (float*)d_out;                  // [2,64]

  // rows = B*C, chunks chosen so every chunk is a multiple of 1024 elems.
  fm_reduce_partial<<<dim3(128, 128), RED_THREADS, 0, stream>>>(x2, ws,         2097152, 128);
  fm_reduce_partial<<<dim3(256,  32), RED_THREADS, 0, stream>>>(x1, ws + 16384,  262144,  32);
  fm_reduce_partial<<<dim3(512,   8), RED_THREADS, 0, stream>>>(x0, ws + 24576,   32768,   8);

  fm_finalize<<<1, 256, 0, stream>>>(ws, ln1w, ln1b, ln2w, ln2b, ln3w, ln3b,
                                     ln4w, ln4b, ln5w, ln5b, w1, b1, w2, b2,
                                     cw, cb, out);
}
